// EncoderBlock_55542517072151
// MI455X (gfx1250) — matrix operand order repaired and checked by reading, not compiled, not run on hardware
//
#include <hip/hip_runtime.h>
#include <hip/hip_bf16.h>

// ---------------------------------------------------------------------------
// Types
// ---------------------------------------------------------------------------
typedef _Float16 v16h __attribute__((ext_vector_type(16)));
typedef _Float16 v8h  __attribute__((ext_vector_type(8)));
typedef float    v8f  __attribute__((ext_vector_type(8)));

// Model constants
#define D_MODEL 1024
#define D_FF    4096
#define NHEAD   16
#define DK_     64
#define SEQ     2048
#define BATCH   2
#define MROWS   (BATCH * SEQ)   // 4096

// ---------------------------------------------------------------------------
// WMMA fragment loaders (CDNA5 CDNA5 ISA layouts, wave32):
//  A (16x32 f16): lane l holds row (l&15); half j -> K = (j>>3)*16 + 8*(l>>4) + (j&7)
//    -> two contiguous 8-half groups at offsets kb and 16+kb (kb = 8*(l>>4)).
//  B (32x16 f16): lane l holds col (l&15); half j -> K = (j>>3)*16 + 8*(l>>4) + (j&7)
//    -> same two 8-half groups as A (loaded via load_fragA).
// ---------------------------------------------------------------------------
__device__ __forceinline__ v16h load_fragA(const _Float16* p /*16B aligned*/) {
    v8h lo = *(const v8h*)p;
    v8h hi = *(const v8h*)(p + 16);
    return __builtin_shufflevector(lo, hi, 0,1,2,3,4,5,6,7,8,9,10,11,12,13,14,15);
}
__device__ __forceinline__ v16h load_fragB(const _Float16* p /*32B aligned*/) {
    return *(const v16h*)p;
}

// ---------------------------------------------------------------------------
// f32 -> f16 elementwise conversion
// ---------------------------------------------------------------------------
__global__ void f32_to_f16_kernel(const float* __restrict__ src,
                                  _Float16* __restrict__ dst, int n) {
    int i = blockIdx.x * blockDim.x + threadIdx.x;
    if (i < n) dst[i] = (_Float16)src[i];
}

// ---------------------------------------------------------------------------
// LayerNorm (faithful: unbiased var ddof=1, alpha*(x-mean)/(std+eps)+beta),
// scalar alpha/beta.  One block per row of 1024 floats.  Output f16.
// ---------------------------------------------------------------------------
__global__ __launch_bounds__(256)
void ln_kernel(const float* __restrict__ x, const float* __restrict__ alpha,
               const float* __restrict__ beta, _Float16* __restrict__ out) {
    __shared__ float ssum[256];
    __shared__ float ssq[256];
    const int row = blockIdx.x, tid = threadIdx.x;
    const float* xr = x + (size_t)row * D_MODEL + tid * 4;
    float v0 = xr[0], v1 = xr[1], v2 = xr[2], v3 = xr[3];
    ssum[tid] = v0 + v1 + v2 + v3;
    ssq[tid]  = v0*v0 + v1*v1 + v2*v2 + v3*v3;
    __syncthreads();
    for (int off = 128; off > 0; off >>= 1) {
        if (tid < off) { ssum[tid] += ssum[tid + off]; ssq[tid] += ssq[tid + off]; }
        __syncthreads();
    }
    const float mean = ssum[0] * (1.0f / D_MODEL);
    const float var  = (ssq[0] - (float)D_MODEL * mean * mean) * (1.0f / (D_MODEL - 1));
    const float stdv = sqrtf(fmaxf(var, 0.0f));
    const float sc   = alpha[0] / (stdv + 1e-6f);
    const float b0   = beta[0];
    _Float16* orow = out + (size_t)row * D_MODEL + tid * 4;
    orow[0] = (_Float16)((v0 - mean) * sc + b0);
    orow[1] = (_Float16)((v1 - mean) * sc + b0);
    orow[2] = (_Float16)((v2 - mean) * sc + b0);
    orow[3] = (_Float16)((v3 - mean) * sc + b0);
}

// ---------------------------------------------------------------------------
// Tiled f16 GEMM with fp32 accumulation via v_wmma_f32_16x16x32_f16.
//   C[M,N] = A[M,K] @ B[K,N] + bias[N]   (+ReLU) (+res[M,N])
// Block tile 128x128x32, 256 threads = 8 waves, wave w owns tile-row w
// (rows 16w..16w+15) x 8 column tiles.  B staged transposed in LDS so the
// B fragment is two contiguous 16B chunks per lane, read like A.
// ---------------------------------------------------------------------------
#define BM 128
#define BN 128
#define BK 32

template<bool RELU, bool RES, bool OUT16>
__global__ __launch_bounds__(256)
void gemm_kernel(const _Float16* __restrict__ A, const _Float16* __restrict__ Bm,
                 const float* __restrict__ bias, const float* __restrict__ res,
                 float* __restrict__ outF, _Float16* __restrict__ outH,
                 int M, int N, int K) {
    __shared__ __align__(16) _Float16 As[BM * BK];   // [m][k], 8 KB
    __shared__ __align__(16) _Float16 Bst[BN * BK];  // transposed [n][k], 8 KB

    const int tid  = threadIdx.x;
    const int wave = tid >> 5;
    const int lane = tid & 31;
    const int hi   = lane >> 4;
    const int ln   = lane & 15;
    const int kb   = hi * 8;
    const int m0   = blockIdx.y * BM;
    const int n0   = blockIdx.x * BN;

    v8f acc[8] = {};

    for (int k0 = 0; k0 < K; k0 += BK) {
        __syncthreads();
        // --- A tile: 128 rows x 32 halves; 2 threads per row, 16 halves each
        {
            const int r = tid >> 1, cs = (tid & 1) * 16;
            const v8h* src = (const v8h*)(A + (size_t)(m0 + r) * K + k0 + cs);
            v8h* dst = (v8h*)(As + r * BK + cs);
            dst[0] = src[0];
            dst[1] = src[1];
        }
        // --- B tile: 32 k-rows x 128 cols, stored transposed into Bst[n][k]
        {
            const int kr = tid >> 3;            // 0..31
            const int cs = (tid & 7) * 16;      // 0..112
            const v8h* src = (const v8h*)(Bm + (size_t)(k0 + kr) * N + n0 + cs);
            v8h t0 = src[0], t1 = src[1];
            #pragma unroll
            for (int i = 0; i < 8; ++i) Bst[(cs + i) * BK + kr] = t0[i];
            #pragma unroll
            for (int i = 0; i < 8; ++i) Bst[(cs + 8 + i) * BK + kr] = t1[i];
        }
        __syncthreads();

        const v16h a = load_fragA(As + (16 * wave + ln) * BK + kb);
        #pragma unroll
        for (int c = 0; c < 8; ++c) {
            const v16h b = load_fragA(Bst + (16 * c + ln) * BK + kb);
            acc[c] = __builtin_amdgcn_wmma_f32_16x16x32_f16(
                false, a, false, b, (short)0, acc[c], false, false);
        }
    }

    // --- epilogue: C layout -> lane holds col (lane&15), VGPR r holds row r+8*hi
    #pragma unroll
    for (int c = 0; c < 8; ++c) {
        const int col = n0 + 16 * c + ln;
        const float bv = bias[col];
        #pragma unroll
        for (int r = 0; r < 8; ++r) {
            const int row = m0 + 16 * wave + r + 8 * hi;
            float v = acc[c][r] + bv;
            if (RELU) v = v > 0.0f ? v : 0.0f;
            if (RES)  v += res[(size_t)row * N + col];
            if (OUT16) outH[(size_t)row * N + col] = (_Float16)v;
            else       outF[(size_t)row * N + col] = v;
        }
    }
}

// ---------------------------------------------------------------------------
// Flash attention (no 1/sqrt(dk) scaling, per the reference source).
// One workgroup (8 waves) per (b, h, 128 query rows).  Each wave owns a
// 16-row q tile.  Streams 64-key blocks: scores via 8 WMMAs, fp32 online
// softmax (shfl_xor reductions within 16-lane halves, matching the C-matrix
// layout), P re-laid-out through LDS (explicit s_wait_dscnt for the
// same-wave RAW), then P@V via 8 WMMAs against an LDS-transposed V tile.
// ---------------------------------------------------------------------------
#define AQ 128   // q rows per workgroup
#define AKB 64   // keys per block iteration

__global__ __launch_bounds__(256)
void attn_kernel(const _Float16* __restrict__ Q, const _Float16* __restrict__ Kt,
                 const _Float16* __restrict__ V, _Float16* __restrict__ Out) {
    __shared__ __align__(16) _Float16 Qs[AQ * DK_];     // 16 KB [q][d]
    __shared__ __align__(16) _Float16 Ks[AKB * DK_];    //  8 KB [key][d]
    __shared__ __align__(16) _Float16 Vst[DK_ * AKB];   //  8 KB [d][key]
    __shared__ __align__(16) _Float16 Ps[8][16 * AKB];  // 16 KB per-wave P tiles

    const int tid  = threadIdx.x;
    const int wave = tid >> 5;
    const int lane = tid & 31;
    const int hi   = lane >> 4;
    const int ln   = lane & 15;
    const int kb   = hi * 8;

    const int qb = blockIdx.x & 15;        // S/128 = 16 q-blocks
    const int bh = blockIdx.x >> 4;        // 0..31
    const int b  = bh >> 4;                // H = 16
    const int h  = bh & 15;
    const size_t rowBase = (size_t)b * SEQ;
    const int col0 = h * DK_;

    // --- load this block's 128 x 64 Q tile
    {
        const int r = tid >> 1, seg = (tid & 1) * 32;
        const v8h* src = (const v8h*)(Q + (rowBase + qb * AQ + r) * D_MODEL + col0 + seg);
        v8h* dst = (v8h*)(Qs + r * DK_ + seg);
        dst[0] = src[0]; dst[1] = src[1]; dst[2] = src[2]; dst[3] = src[3];
    }

    v8f o[4] = {};                  // 16 q-rows x 64 d, fp32 accum
    float rm[8], rl[8];
    #pragma unroll
    for (int r = 0; r < 8; ++r) { rm[r] = -1.0e30f; rl[r] = 0.0f; }

    for (int k0 = 0; k0 < SEQ; k0 += AKB) {
        __syncthreads();
        // --- cooperative load: K tile [64][64] and V transposed into [d][key]
        {
            const int kr = tid >> 2;           // 0..63
            const int cs = (tid & 3) * 16;     // 0,16,32,48
            const v8h* ksrc = (const v8h*)(Kt + (rowBase + k0 + kr) * D_MODEL + col0 + cs);
            v8h* kdst = (v8h*)(Ks + kr * DK_ + cs);
            kdst[0] = ksrc[0]; kdst[1] = ksrc[1];
            const v8h* vsrc = (const v8h*)(V + (rowBase + k0 + kr) * D_MODEL + col0 + cs);
            v8h t0 = vsrc[0], t1 = vsrc[1];
            #pragma unroll
            for (int i = 0; i < 8; ++i) Vst[(cs + i) * AKB + kr] = t0[i];
            #pragma unroll
            for (int i = 0; i < 8; ++i) Vst[(cs + 8 + i) * AKB + kr] = t1[i];
        }
        __syncthreads();

        // --- scores S = Q . K^T  (M=16 q, N=64 keys, Kdim=64 -> 2 k-steps)
        v8f sc[4] = {};
        #pragma unroll
        for (int ks = 0; ks < 2; ++ks) {
            const v16h a = load_fragA(Qs + (16 * wave + ln) * DK_ + ks * 32 + kb);
            #pragma unroll
            for (int c = 0; c < 4; ++c) {
                const v16h bm = load_fragA(Ks + (16 * c + ln) * DK_ + ks * 32 + kb);
                sc[c] = __builtin_amdgcn_wmma_f32_16x16x32_f16(
                    false, a, false, bm, (short)0, sc[c], false, false);
            }
        }

        // --- fp32 online softmax over this 64-key block
        #pragma unroll
        for (int r = 0; r < 8; ++r) {
            const float s0 = sc[0][r], s1 = sc[1][r], s2 = sc[2][r], s3 = sc[3][r];
            float mx = fmaxf(fmaxf(s0, s1), fmaxf(s2, s3));
            #pragma unroll
            for (int m = 1; m < 16; m <<= 1)
                mx = fmaxf(mx, __shfl_xor(mx, m, 32));   // stays in 16-lane half
            const float mnew  = fmaxf(rm[r], mx);
            const float scale = __expf(rm[r] - mnew);
            const float p0 = __expf(s0 - mnew);
            const float p1 = __expf(s1 - mnew);
            const float p2 = __expf(s2 - mnew);
            const float p3 = __expf(s3 - mnew);
            float ps = (p0 + p1) + (p2 + p3);
            #pragma unroll
            for (int m = 1; m < 16; m <<= 1)
                ps += __shfl_xor(ps, m, 32);
            rl[r] = rl[r] * scale + ps;
            rm[r] = mnew;
            #pragma unroll
            for (int c = 0; c < 4; ++c) o[c][r] *= scale;
            const int ql = r + 8 * hi;                   // C-layout row
            Ps[wave][ql * AKB + ln]      = (_Float16)p0;
            Ps[wave][ql * AKB + 16 + ln] = (_Float16)p1;
            Ps[wave][ql * AKB + 32 + ln] = (_Float16)p2;
            Ps[wave][ql * AKB + 48 + ln] = (_Float16)p3;
        }
        // Same-wave LDS RAW: be explicit before re-reading Ps as A fragments.
        asm volatile("s_wait_dscnt 0" ::: "memory");

        // --- O += P @ V  (M=16 q, N=64 d, Kdim=64 keys -> 2 k-steps)
        #pragma unroll
        for (int ks = 0; ks < 2; ++ks) {
            const v16h pa = load_fragA(&Ps[wave][ln * AKB + ks * 32 + kb]);
            #pragma unroll
            for (int c = 0; c < 4; ++c) {
                const v16h bm = load_fragA(Vst + (16 * c + ln) * AKB + ks * 32 + kb);
                o[c] = __builtin_amdgcn_wmma_f32_16x16x32_f16(
                    false, pa, false, bm, (short)0, o[c], false, false);
            }
        }
    }

    // --- normalize and store (f16, laid out [b*S+s][h*64+d])
    #pragma unroll
    for (int c = 0; c < 4; ++c) {
        const int d = 16 * c + ln;
        #pragma unroll
        for (int r = 0; r < 8; ++r) {
            const int qrow = qb * AQ + 16 * wave + r + 8 * hi;
            const float v = o[c][r] / rl[r];
            Out[(rowBase + qrow) * D_MODEL + col0 + d] = (_Float16)v;
        }
    }
}

// ---------------------------------------------------------------------------
// Host launcher
// ---------------------------------------------------------------------------
extern "C" void kernel_launch(void* const* d_in, const int* in_sizes, int n_in,
                              void* d_out, int out_size, void* d_ws, size_t ws_size,
                              hipStream_t stream) {
    (void)in_sizes; (void)n_in; (void)out_size; (void)ws_size;

    const float* x    = (const float*)d_in[0];
    const float* Wq   = (const float*)d_in[1];
    const float* bq   = (const float*)d_in[2];
    const float* Wk   = (const float*)d_in[3];
    const float* bk   = (const float*)d_in[4];
    const float* Wv   = (const float*)d_in[5];
    const float* bv   = (const float*)d_in[6];
    const float* Wo   = (const float*)d_in[7];
    const float* bo   = (const float*)d_in[8];
    const float* W1   = (const float*)d_in[9];
    const float* b1   = (const float*)d_in[10];
    const float* W2   = (const float*)d_in[11];
    const float* b2   = (const float*)d_in[12];
    const float* ln1a = (const float*)d_in[13];
    const float* ln1b = (const float*)d_in[14];
    const float* ln2a = (const float*)d_in[15];
    const float* ln2b = (const float*)d_in[16];
    float* outp = (float*)d_out;

    char* ws = (char*)d_ws;
    const size_t MB = 1024u * 1024u;
    _Float16* wq16  = (_Float16*)(ws + 0 * MB);
    _Float16* wk16  = (_Float16*)(ws + 2 * MB);
    _Float16* wv16  = (_Float16*)(ws + 4 * MB);
    _Float16* wo16  = (_Float16*)(ws + 6 * MB);
    _Float16* w116  = (_Float16*)(ws + 8 * MB);
    _Float16* w216  = (_Float16*)(ws + 16 * MB);
    _Float16* n16   = (_Float16*)(ws + 24 * MB);   // LN1 output, reused for LN2
    _Float16* q16   = (_Float16*)(ws + 32 * MB);
    _Float16* k16   = (_Float16*)(ws + 40 * MB);
    _Float16* v16   = (_Float16*)(ws + 48 * MB);
    _Float16* at16  = (_Float16*)(ws + 56 * MB);
    float*    hbuf  = (float*)   (ws + 64 * MB);   // fp32 residual h (16 MB)
    _Float16* ffn16 = (_Float16*)(ws + 32 * MB);   // reuses q/k/v/attn region

    const int NSMALL = D_MODEL * D_MODEL;   // 1048576
    const int NBIG   = D_MODEL * D_FF;      // 4194304

    // 1) weight conversions f32 -> f16
    f32_to_f16_kernel<<<(NSMALL + 255) / 256, 256, 0, stream>>>(Wq, wq16, NSMALL);
    f32_to_f16_kernel<<<(NSMALL + 255) / 256, 256, 0, stream>>>(Wk, wk16, NSMALL);
    f32_to_f16_kernel<<<(NSMALL + 255) / 256, 256, 0, stream>>>(Wv, wv16, NSMALL);
    f32_to_f16_kernel<<<(NSMALL + 255) / 256, 256, 0, stream>>>(Wo, wo16, NSMALL);
    f32_to_f16_kernel<<<(NBIG   + 255) / 256, 256, 0, stream>>>(W1, w116, NBIG);
    f32_to_f16_kernel<<<(NBIG   + 255) / 256, 256, 0, stream>>>(W2, w216, NBIG);

    // 2) LN1
    ln_kernel<<<MROWS, 256, 0, stream>>>(x, ln1a, ln1b, n16);

    // 3) Q/K/V projections
    dim3 gQKV(D_MODEL / BN, MROWS / BM);   // (8, 32)
    gemm_kernel<false, false, true><<<gQKV, 256, 0, stream>>>(
        n16, wq16, bq, nullptr, nullptr, q16, MROWS, D_MODEL, D_MODEL);
    gemm_kernel<false, false, true><<<gQKV, 256, 0, stream>>>(
        n16, wk16, bk, nullptr, nullptr, k16, MROWS, D_MODEL, D_MODEL);
    gemm_kernel<false, false, true><<<gQKV, 256, 0, stream>>>(
        n16, wv16, bv, nullptr, nullptr, v16, MROWS, D_MODEL, D_MODEL);

    // 4) flash attention: B*H*(S/128) = 512 workgroups
    attn_kernel<<<BATCH * NHEAD * (SEQ / AQ), 256, 0, stream>>>(q16, k16, v16, at16);

    // 5) output projection + residual(x) -> h (fp32)
    gemm_kernel<false, true, false><<<gQKV, 256, 0, stream>>>(
        at16, wo16, bo, x, hbuf, nullptr, MROWS, D_MODEL, D_MODEL);

    // 6) LN2
    ln_kernel<<<MROWS, 256, 0, stream>>>(hbuf, ln2a, ln2b, n16);

    // 7) FFN1 with ReLU -> f16
    dim3 gFF1(D_FF / BN, MROWS / BM);      // (32, 32)
    gemm_kernel<true, false, true><<<gFF1, 256, 0, stream>>>(
        n16, w116, b1, nullptr, nullptr, ffn16, MROWS, D_FF, D_MODEL);

    // 8) FFN2 + residual(h) -> fp32 output
    gemm_kernel<false, true, false><<<gQKV, 256, 0, stream>>>(
        ffn16, w216, b2, hbuf, outp, nullptr, MROWS, D_MODEL, D_FF);
}
